// GCNFN_77850577207793
// MI455X (gfx1250) — compile-verified, matrix operand order checked by reading, not run on hardware
//
#include <hip/hip_runtime.h>
#include <hip/hip_bf16.h>
#include <math.h>

typedef __attribute__((ext_vector_type(16))) __bf16 v16bf;
typedef __attribute__((ext_vector_type(8)))  float  v8f;

#define C_DIM 256
#define BM 128
#define BN 128
#define BK 32
#define LDK 40    // A tile LDS row stride (bf16): 80B rows, 16B-aligned
#define LDB 328   // B panel LDS row stride (bf16): 656B rows, 16B-aligned, bank-skewed
#define KP_MAX 320

__device__ __forceinline__ float selu_f(float x) {
    const float a = 1.6732632423543772f, s = 1.0507009873554805f;
    return x > 0.f ? s * x : s * a * (expf(x) - 1.f);
}

// ---------------------------------------------------------------------------
// Pre-transpose + convert weights: Bt[n*Kp + k] = bf16(B[k*N + n]), zero-pad
// k in [K, Kp). Tiny (<=320x256), runs once per GEMM.
// ---------------------------------------------------------------------------
__global__ void prep_bt_kernel(const float* __restrict__ B, __bf16* __restrict__ Bt,
                               int K, int N, int Kp)
{
    const int i = blockIdx.x * blockDim.x + threadIdx.x;
    if (i >= N * Kp) return;
    const int n = i / Kp, k = i - n * Kp;
    Bt[i] = (k < K) ? (__bf16)B[(long long)k * N + n] : (__bf16)0.f;
}

// ---------------------------------------------------------------------------
// WMMA GEMM: C[M,N] = A[M,K] @ B[K,N]   (A fp32, B pre-transposed bf16 [N][Kp])
// 128x128 block tile, 8 waves (4 row-groups x 2 col-groups), wave = 32x64.
// B panel resident in LDS for whole kernel; A tile double-buffered with one
// barrier per K-step; next A tile's global loads overlap WMMA.
// ---------------------------------------------------------------------------
__global__ __launch_bounds__(256)
void wmma_gemm_kernel(const float* __restrict__ A, const __bf16* __restrict__ Bt,
                      float* __restrict__ Cout, int M, int K, int N, int Kp)
{
    extern __shared__ __align__(16) char smem[];
    __bf16* sA0 = (__bf16*)smem;                 // BM*LDK
    __bf16* sA1 = sA0 + BM * LDK;                // BM*LDK
    __bf16* sB  = sA1 + BM * LDK;                // BN*LDB (persistent panel)

    const int tid  = threadIdx.x;
    const int wid  = tid >> 5;
    const int lane = tid & 31;
    const int wm   = wid & 3;     // 0..3  -> 32-row group
    const int wn   = wid >> 2;    // 0..1  -> 64-col group
    const int m0   = blockIdx.x * BM;
    const int n0   = blockIdx.y * BN;

    // staging assignments (per thread: 16 contiguous elements per 128x32 tile)
    const int arow = tid >> 1;
    const int akb  = (tid & 1) * 16;
    const int agr  = m0 + arow;
    const int brow = tid >> 1;
    const int bkb  = (tid & 1) * 16;

    const bool mfull = (m0 + BM <= M);
    const bool k4al  = ((K & 3) == 0);   // A rows 16B-aligned

    // ---- preload the full B panel for this block into LDS (once) ----
    for (int k0 = 0; k0 < Kp; k0 += BK) {
        const __bf16* bb = Bt + (long long)(n0 + brow) * Kp + k0 + bkb;
        const uint4 u0 = *(const uint4*)bb;
        const uint4 u1 = *(const uint4*)(bb + 8);
        *(uint4*)&sB[brow * LDB + k0 + bkb]     = u0;
        *(uint4*)&sB[brow * LDB + k0 + bkb + 8] = u1;
    }

    float aR[16];

    auto load_A = [&](int k0) {
        const float* abase = A + (long long)agr * K + k0 + akb;
        if (mfull && (k0 + BK <= K)) {
            if (k4al) {
                const float4* av = (const float4*)abase;
                #pragma unroll
                for (int q = 0; q < 4; ++q) {
                    const float4 v = av[q];
                    aR[q * 4 + 0] = v.x; aR[q * 4 + 1] = v.y;
                    aR[q * 4 + 2] = v.z; aR[q * 4 + 3] = v.w;
                }
            } else {
                #pragma unroll
                for (int i = 0; i < 16; ++i) aR[i] = abase[i];
            }
        } else {
            #pragma unroll
            for (int i = 0; i < 16; ++i)
                aR[i] = (agr < M && (k0 + akb + i) < K) ? abase[i] : 0.f;
        }
    };

    auto store_A = [&](__bf16* buf) {
        union { __bf16 h[16]; uint4 u[2]; } pk;
        #pragma unroll
        for (int i = 0; i < 16; ++i) pk.h[i] = (__bf16)aR[i];
        *(uint4*)&buf[arow * LDK + akb]     = pk.u[0];
        *(uint4*)&buf[arow * LDK + akb + 8] = pk.u[1];
    };

    v8f acc[2][4];
    #pragma unroll
    for (int i = 0; i < 2; ++i)
        #pragma unroll
        for (int j = 0; j < 4; ++j)
            acc[i][j] = {};

    const int lrow = lane & 15;
    const int kh   = (lane >> 4) * 8;   // K-half select per ISA 16-bit A/B layout

    const int steps = Kp / BK;

    // prologue: stage A tile 0 into buffer 0 (also fences the B panel preload)
    load_A(0);
    store_A(sA0);
    __syncthreads();

    for (int s = 0; s < steps; ++s) {
        __bf16* cur = (s & 1) ? sA1 : sA0;
        __bf16* nxt = (s & 1) ? sA0 : sA1;

        if (s + 1 < steps) load_A((s + 1) * BK);   // global loads overlap WMMA

        union Frag { v16bf v; uint4 u[2]; };
        Frag fa[2], fb[4];
        #pragma unroll
        for (int i = 0; i < 2; ++i) {
            const __bf16* p = &cur[(wm * 32 + i * 16 + lrow) * LDK];
            fa[i].u[0] = *(const uint4*)(p + kh);        // K 0..7   (+kh)
            fa[i].u[1] = *(const uint4*)(p + 16 + kh);   // K 16..23 (+kh)
        }
        const int kp0 = s * BK;
        #pragma unroll
        for (int j = 0; j < 4; ++j) {
            const __bf16* p = &sB[(wn * 64 + j * 16 + lrow) * LDB + kp0];
            fb[j].u[0] = *(const uint4*)(p + kh);
            fb[j].u[1] = *(const uint4*)(p + 16 + kh);
        }
        #pragma unroll
        for (int i = 0; i < 2; ++i)
            #pragma unroll
            for (int j = 0; j < 4; ++j)
                acc[i][j] = __builtin_amdgcn_wmma_f32_16x16x32_bf16(
                    false, fa[i].v, false, fb[j].v, (short)0, acc[i][j], false, false);

        if (s + 1 < steps) {
            store_A(nxt);       // safe: nxt's old contents last read before prev barrier
            __syncthreads();
        }
    }

    // ---- epilogue: C/D layout — VGPR r, lanes 0-15: M=r, lanes 16-31: M=8+r ----
    const int lcol  = lane & 15;
    const int rquad = (lane >> 4) * 8;
    #pragma unroll
    for (int i = 0; i < 2; ++i) {
        #pragma unroll
        for (int j = 0; j < 4; ++j) {
            const int col = n0 + wn * 64 + j * 16 + lcol;
            #pragma unroll
            for (int r = 0; r < 8; ++r) {
                const int row = m0 + wm * 32 + i * 16 + rquad + r;
                if (row < M) Cout[(long long)row * N + col] = acc[i][j][r];
            }
        }
    }
}

// ---------------------------------------------------------------------------
// a_src[n] = h[n,:].att_l ; a_dst[n] = h[n,:].att_r   (wave32 per node,
// 8 contiguous channels per lane -> 2x global_load_b128 per lane)
// ---------------------------------------------------------------------------
__global__ __launch_bounds__(256)
void attn_scores_kernel(const float* __restrict__ h, const float* __restrict__ attl,
                        const float* __restrict__ attr, float* __restrict__ asrc,
                        float* __restrict__ adst, int Nn)
{
    const int lane = threadIdx.x & 31;
    const long long node = (long long)blockIdx.x * 8 + (threadIdx.x >> 5);
    if (node >= Nn) return;
    const float4* hv = (const float4*)(h + node * C_DIM + lane * 8);
    const float4* lv = (const float4*)(attl + lane * 8);
    const float4* rv = (const float4*)(attr + lane * 8);
    float sl = 0.f, sr = 0.f;
    #pragma unroll
    for (int q = 0; q < 2; ++q) {
        const float4 hh = hv[q], ll = lv[q], rr = rv[q];
        sl += hh.x * ll.x + hh.y * ll.y + hh.z * ll.z + hh.w * ll.w;
        sr += hh.x * rr.x + hh.y * rr.y + hh.z * rr.z + hh.w * rr.w;
    }
    #pragma unroll
    for (int off = 16; off > 0; off >>= 1) {
        sl += __shfl_down(sl, off, 32);
        sr += __shfl_down(sr, off, 32);
    }
    if (lane == 0) { asrc[node] = sl; adst[node] = sr; }
}

// ---------------------------------------------------------------------------
__global__ void init_layer_kernel(float* m, float* denom, float* agg,
                                  const float* __restrict__ bias, int Nn)
{
    const long long i = (long long)blockIdx.x * blockDim.x + threadIdx.x;
    const long long tot = (long long)Nn * C_DIM;
    if (i < tot) agg[i] = bias[i & (C_DIM - 1)];
    if (i < Nn) { m[i] = -__builtin_inff(); denom[i] = 0.f; }
}

__global__ void zero_kernel(float* p, long long n)
{
    const long long i = (long long)blockIdx.x * blockDim.x + threadIdx.x;
    if (i < n) p[i] = 0.f;
}

__device__ __forceinline__ void atomicMaxF(float* addr, float v)
{
    if (v >= 0.f) atomicMax((int*)addr, __float_as_int(v));
    else          atomicMin((unsigned int*)addr, (unsigned int)__float_as_int(v));
}

// ---------------------------------------------------------------------------
// Pass 1: e = leaky_relu(a_src[s]+a_dst[d]); segment max into m[d]
// ---------------------------------------------------------------------------
__global__ void edge_max_kernel(const int* __restrict__ src, const int* __restrict__ dst,
                                long long E, int Nn,
                                const float* __restrict__ asrc, const float* __restrict__ adst,
                                float* __restrict__ ebuf, float* __restrict__ m)
{
    const long long i = (long long)blockIdx.x * blockDim.x + threadIdx.x;
    const long long Etot = E + Nn;
    if (i >= Etot) return;
    const int s = (i < E) ? src[i] : (int)(i - E);   // appended self-loops
    const int d = (i < E) ? dst[i] : (int)(i - E);
    float e = asrc[s] + adst[d];
    e = e > 0.f ? e : 0.2f * e;
    ebuf[i] = e;
    atomicMaxF(&m[d], e);
}

// ---------------------------------------------------------------------------
// Pass 2: ex = exp(e - m[d]); segment sum into denom[d]
// ---------------------------------------------------------------------------
__global__ void edge_expsum_kernel(const int* __restrict__ src, const int* __restrict__ dst,
                                   long long E, int Nn,
                                   const float* __restrict__ m,
                                   float* __restrict__ ebuf, float* __restrict__ denom)
{
    const long long i = (long long)blockIdx.x * blockDim.x + threadIdx.x;
    const long long Etot = E + Nn;
    if (i >= Etot) return;
    const int d = (i < E) ? dst[i] : (int)(i - E);
    float mv = m[d];
    if (!(mv > -3.0e38f)) mv = 0.f;   // where(isfinite(m), m, 0)
    const float ex = expf(ebuf[i] - mv);
    ebuf[i] = ex;
    atomicAdd(&denom[d], ex);
}

// ---------------------------------------------------------------------------
// Pass 3: agg[d,:] += (ex/denom[d]) * h[s,:]   (wave32 per edge; each lane
// owns 8 contiguous channels -> 2x global_load_b128 gather + 8 f32 atomics)
// ---------------------------------------------------------------------------
__global__ __launch_bounds__(256)
void edge_aggregate_kernel(const int* __restrict__ src, const int* __restrict__ dst,
                           long long E, int Nn,
                           const float* __restrict__ ebuf, const float* __restrict__ denom,
                           const float* __restrict__ h, float* __restrict__ agg)
{
    const int lane = threadIdx.x & 31;
    const long long edge = (long long)blockIdx.x * 8 + (threadIdx.x >> 5);
    const long long Etot = E + Nn;
    if (edge >= Etot) return;
    const int s = (edge < E) ? src[edge] : (int)(edge - E);
    const int d = (edge < E) ? dst[edge] : (int)(edge - E);
    const float alpha = ebuf[edge] / fmaxf(denom[d], 1e-16f);
    const float4* hv = (const float4*)(h + (long long)s * C_DIM + lane * 8);
    float*        od = agg + (long long)d * C_DIM + lane * 8;
    const float4 v0 = hv[0], v1 = hv[1];
    atomicAdd(&od[0], alpha * v0.x);
    atomicAdd(&od[1], alpha * v0.y);
    atomicAdd(&od[2], alpha * v0.z);
    atomicAdd(&od[3], alpha * v0.w);
    atomicAdd(&od[4], alpha * v1.x);
    atomicAdd(&od[5], alpha * v1.y);
    atomicAdd(&od[6], alpha * v1.z);
    atomicAdd(&od[7], alpha * v1.w);
}

__global__ void selu_kernel(float* p, long long n)
{
    const long long i = (long long)blockIdx.x * blockDim.x + threadIdx.x;
    if (i < n) p[i] = selu_f(p[i]);
}

// ---------------------------------------------------------------------------
// Mean pool over graphs (wave32 per node, 8 contiguous channels per lane)
// ---------------------------------------------------------------------------
__global__ __launch_bounds__(256)
void pool_kernel(const float* __restrict__ h, const int* __restrict__ batch,
                 float* __restrict__ gsum, float* __restrict__ gcnt, int Nn)
{
    const int lane = threadIdx.x & 31;
    const long long node = (long long)blockIdx.x * 8 + (threadIdx.x >> 5);
    if (node >= Nn) return;
    const int g = batch[node];
    const float4* hv = (const float4*)(h + node * C_DIM + lane * 8);
    float*        os = gsum + (long long)g * C_DIM + lane * 8;
    const float4 v0 = hv[0], v1 = hv[1];
    atomicAdd(&os[0], v0.x); atomicAdd(&os[1], v0.y);
    atomicAdd(&os[2], v0.z); atomicAdd(&os[3], v0.w);
    atomicAdd(&os[4], v1.x); atomicAdd(&os[5], v1.y);
    atomicAdd(&os[6], v1.z); atomicAdd(&os[7], v1.w);
    if (lane == 0) atomicAdd(&gcnt[g], 1.f);
}

__global__ void pool_finalize_kernel(const float* __restrict__ gsum,
                                     const float* __restrict__ gcnt,
                                     float* __restrict__ gvec, int G)
{
    const int i = blockIdx.x * blockDim.x + threadIdx.x;
    if (i < G * C_DIM) {
        const int g = i / C_DIM;
        gvec[i] = selu_f(gsum[i] / fmaxf(gcnt[g], 1.f));
    }
}

__global__ void fc1_kernel(const float* __restrict__ g, const float* __restrict__ W,
                           const float* __restrict__ b, float* __restrict__ out,
                           int G, int H)
{
    const int i = blockIdx.x * blockDim.x + threadIdx.x;
    if (i >= G * H) return;
    const int gi = i / H, hh = i - gi * H;
    float s = b[hh];
    for (int c = 0; c < C_DIM; ++c) s += g[gi * C_DIM + c] * W[c * H + hh];
    out[i] = selu_f(s);
}

__global__ void fc2_lsm_kernel(const float* __restrict__ f, const float* __restrict__ W,
                               const float* __restrict__ b, float* __restrict__ out,
                               int G, int H)
{
    const int gi = blockIdx.x * blockDim.x + threadIdx.x;
    if (gi >= G) return;
    float z0 = b[0], z1 = b[1];
    for (int c = 0; c < H; ++c) {
        const float v = f[gi * H + c];
        z0 += v * W[c * 2 + 0];
        z1 += v * W[c * 2 + 1];
    }
    const float mx  = fmaxf(z0, z1);
    const float lse = mx + logf(expf(z0 - mx) + expf(z1 - mx));
    out[gi * 2 + 0] = z0 - lse;
    out[gi * 2 + 1] = z1 - lse;
}

// ---------------------------------------------------------------------------
extern "C" void kernel_launch(void* const* d_in, const int* in_sizes, int n_in,
                              void* d_out, int out_size, void* d_ws, size_t ws_size,
                              hipStream_t stream)
{
    (void)n_in; (void)ws_size;
    // inputs: x, edge_index, batch, W1, att_l1, att_r1, b1, W2, att_l2, att_r2, b2,
    //         fc1_W, fc1_b, fc2_W, fc2_b
    const float* x      = (const float*)d_in[0];
    const int*   eidx   = (const int*)d_in[1];
    const int*   batch  = (const int*)d_in[2];
    const float* W1     = (const float*)d_in[3];
    const float* attl1  = (const float*)d_in[4];
    const float* attr1  = (const float*)d_in[5];
    const float* b1     = (const float*)d_in[6];
    const float* W2     = (const float*)d_in[7];
    const float* attl2  = (const float*)d_in[8];
    const float* attr2  = (const float*)d_in[9];
    const float* b2     = (const float*)d_in[10];
    const float* fc1W   = (const float*)d_in[11];
    const float* fc1b   = (const float*)d_in[12];
    const float* fc2W   = (const float*)d_in[13];
    const float* fc2b   = (const float*)d_in[14];
    float* out = (float*)d_out;

    const int Nn   = in_sizes[2];               // 50000 nodes
    const int Fin  = in_sizes[0] / Nn;          // 310
    const long long E = in_sizes[1] / 2;        // 800000 edges
    const int H    = in_sizes[12];              // 128
    const int G    = out_size / 2;              // 256 graphs
    const int C    = C_DIM;                     // 256
    const long long Etot = E + Nn;
    const long long NC   = (long long)Nn * C;
    const int Kp1  = ((Fin + BK - 1) / BK) * BK;   // 320
    const int Kp2  = C;                            // 256 (already mult of 32)

    const int* src = eidx;
    const int* dst = eidx + E;

    // workspace carve-up
    char* w = (char*)d_ws;
    auto alloc = [&](size_t bytes) -> void* {
        void* p = (void*)w;
        w += (bytes + 255) & ~(size_t)255;
        return p;
    };
    float*  bufA  = (float*)alloc(NC * 4);           // GEMM outputs (h)
    float*  bufB  = (float*)alloc(NC * 4);           // aggregation outputs
    float*  asrc  = (float*)alloc((size_t)Nn * 4);
    float*  adst  = (float*)alloc((size_t)Nn * 4);
    float*  mmax  = (float*)alloc((size_t)Nn * 4);
    float*  denom = (float*)alloc((size_t)Nn * 4);
    float*  ebuf  = (float*)alloc((size_t)Etot * 4);
    float*  gsum  = (float*)alloc((size_t)G * C * 4);
    float*  gcnt  = (float*)alloc((size_t)G * 4);
    float*  gvec  = (float*)alloc((size_t)G * C * 4);
    float*  fc1o  = (float*)alloc((size_t)G * H * 4);
    __bf16* Bt    = (__bf16*)alloc((size_t)C * KP_MAX * 2);

    const dim3 blk(256);
    const dim3 gemm_grid((Nn + BM - 1) / BM, C / BN);
    const size_t gemm_smem = (size_t)(2 * BM * LDK + BN * LDB) * 2;   // ~102 KB
    const int  nc_blocks    = (int)((NC + 255) / 256);
    const int  edge_blocks  = (int)((Etot + 255) / 256);
    const int  edgew_blocks = (int)((Etot + 7) / 8);
    const int  nodew_blocks = (Nn + 7) / 8;

    // ===================== Layer 1 =====================
    prep_bt_kernel<<<(C * Kp1 + 255) / 256, blk, 0, stream>>>(W1, Bt, Fin, C, Kp1);
    wmma_gemm_kernel<<<gemm_grid, blk, gemm_smem, stream>>>(x, Bt, bufA, Nn, Fin, C, Kp1);
    attn_scores_kernel<<<nodew_blocks, blk, 0, stream>>>(bufA, attl1, attr1, asrc, adst, Nn);
    init_layer_kernel<<<nc_blocks, blk, 0, stream>>>(mmax, denom, bufB, b1, Nn);
    edge_max_kernel<<<edge_blocks, blk, 0, stream>>>(src, dst, E, Nn, asrc, adst, ebuf, mmax);
    edge_expsum_kernel<<<edge_blocks, blk, 0, stream>>>(src, dst, E, Nn, mmax, ebuf, denom);
    edge_aggregate_kernel<<<edgew_blocks, blk, 0, stream>>>(src, dst, E, Nn, ebuf, denom, bufA, bufB);
    selu_kernel<<<nc_blocks, blk, 0, stream>>>(bufB, NC);

    // ===================== Layer 2 =====================
    prep_bt_kernel<<<(C * Kp2 + 255) / 256, blk, 0, stream>>>(W2, Bt, C, C, Kp2);
    wmma_gemm_kernel<<<gemm_grid, blk, gemm_smem, stream>>>(bufB, Bt, bufA, Nn, C, C, Kp2);
    attn_scores_kernel<<<nodew_blocks, blk, 0, stream>>>(bufA, attl2, attr2, asrc, adst, Nn);
    init_layer_kernel<<<nc_blocks, blk, 0, stream>>>(mmax, denom, bufB, b2, Nn);
    edge_max_kernel<<<edge_blocks, blk, 0, stream>>>(src, dst, E, Nn, asrc, adst, ebuf, mmax);
    edge_expsum_kernel<<<edge_blocks, blk, 0, stream>>>(src, dst, E, Nn, mmax, ebuf, denom);
    edge_aggregate_kernel<<<edgew_blocks, blk, 0, stream>>>(src, dst, E, Nn, ebuf, denom, bufA, bufB);
    selu_kernel<<<nc_blocks, blk, 0, stream>>>(bufB, NC);

    // ===================== Pool + MLP head =====================
    zero_kernel<<<(int)(((long long)G * C + G + 255) / 256), blk, 0, stream>>>(gsum, (long long)G * C + G);
    pool_kernel<<<nodew_blocks, blk, 0, stream>>>(bufB, batch, gsum, gcnt, Nn);
    pool_finalize_kernel<<<(G * C + 255) / 256, blk, 0, stream>>>(gsum, gcnt, gvec, G);
    fc1_kernel<<<(G * H + 255) / 256, blk, 0, stream>>>(gvec, fc1W, fc1b, fc1o, G, H);
    fc2_lsm_kernel<<<(G + 255) / 256, blk, 0, stream>>>(fc1o, fc2W, fc2b, out, G, H);
}
